// Attention_82325933129950
// MI455X (gfx1250) — matrix operand order repaired and checked by reading, not compiled, not run on hardware
//
#include <hip/hip_runtime.h>

typedef __bf16 bf16_t;
typedef __attribute__((ext_vector_type(16))) __bf16 v16bf;
typedef __attribute__((ext_vector_type(8)))  __bf16 v8bf;
typedef __attribute__((ext_vector_type(8)))  float  v8f;
typedef __attribute__((ext_vector_type(4)))  unsigned int v4u;
typedef __attribute__((ext_vector_type(8)))  int v8i;
typedef __attribute__((ext_vector_type(4)))  int v4i;

#define BATCH 2
#define SEQ   2048
#define DIM   1024
#define HEADS 16
#define DHEAD 64
#define ROWS  (BATCH * SEQ)   // 4096
#define QKVN  (3 * DIM)       // 3072

// ---------------------------------------------------------------------------
// WMMA helpers (CDNA5 v_wmma_f32_16x16x32_bf16, wave32)
// ---------------------------------------------------------------------------
__device__ __forceinline__ v8f wmma_bf16(v16bf a, v16bf b, v8f c) {
  return __builtin_amdgcn_wmma_f32_16x16x32_bf16(false, a, false, b, (short)0, c,
                                                 false, false);
}

// A fragment 16x32 from row-major [row][k] tile: per lane = two contiguous
// 16B runs (K kbase+koff+0..7 and kbase+koff+16..23, koff = 0/8 by lane half).
__device__ __forceinline__ v16bf load_a_frag(const bf16_t* p0, int row0, int stride,
                                             int kbase, int lane) {
  int m = lane & 15;
  int koff = (lane >> 4) << 3;
  const bf16_t* p = p0 + (size_t)(row0 + m) * stride + kbase + koff;
  v8bf lo = *reinterpret_cast<const v8bf*>(p);
  v8bf hi = *reinterpret_cast<const v8bf*>(p + 16);
  return __builtin_shufflevector(lo, hi, 0, 1, 2, 3, 4, 5, 6, 7,
                                 8, 9, 10, 11, 12, 13, 14, 15);
}

// B fragment 32x16 where gemm-B[k][n] = src[n][k] (src row-major [n][k]):
// per lane = two 8-element k runs of row (col0 + lane&15): kbase+koff+0..7 and
// kbase+koff+16..23, koff = 0/8 by lane half.
__device__ __forceinline__ v16bf load_bt_frag(const bf16_t* p0, int col0, int stride,
                                              int kbase, int lane) {
  int n = lane & 15;
  int koff = (lane >> 4) << 3;
  const bf16_t* p = p0 + (size_t)(col0 + n) * stride + kbase + koff;
  v8bf lo = *reinterpret_cast<const v8bf*>(p);
  v8bf hi = *reinterpret_cast<const v8bf*>(p + 16);
  return __builtin_shufflevector(lo, hi, 0, 1, 2, 3, 4, 5, 6, 7,
                                 8, 9, 10, 11, 12, 13, 14, 15);
}

// ---------------------------------------------------------------------------
// TDM: 2D 64x64 bf16 tile load, global (row stride 64 elem) -> LDS with
// 16B padding per 128B row => LDS stride 72 bf16. D# per ISA 8.3/8.4.
// 6-arg toolchain variant: (g0, g1, g2, g3, g4, cpol).
// ---------------------------------------------------------------------------
__device__ __forceinline__ void tdm_load_tile64(const bf16_t* gsrc, unsigned lds_off) {
  unsigned long long ga = (unsigned long long)(size_t)gsrc;
  v4u g0;
  g0[0] = 1u;                                   // count=1 valid descriptor
  g0[1] = lds_off;                              // lds_addr (bytes)
  g0[2] = (unsigned)ga;                         // global_addr[31:0]
  g0[3] = (unsigned)((ga >> 32) & 0x01ffffffu)  // global_addr[56:32]
          | (2u << 30);                         // type = 2 ("image")
  v8i g1;
  // data_size=1(2B), pad_enable, pad_interval code4 (32 DW = 128B),
  // pad_amount code3 (4 DW = 16B)
  g1[0] = (1 << 16) | (1 << 20) | (4 << 22) | (3 << 25);
  g1[1] = (int)(64u << 16);   // tensor_dim0[15:0] = 64
  g1[2] = (int)(64u << 16);   // tensor_dim0[31:16]=0, tensor_dim1[15:0] = 64
  g1[3] = (int)(64u << 16);   // tensor_dim1[31:16]=0, tile_dim0 = 64
  g1[4] = 64;                 // tile_dim1 = 64, tile_dim2 = 0
  g1[5] = 64;                 // tensor_dim0_stride[31:0] = 64 elements
  g1[6] = 0;
  g1[7] = 0;
  v4i z4 = {0, 0, 0, 0};
  v8i z8 = {0, 0, 0, 0, 0, 0, 0, 0};
  __builtin_amdgcn_tensor_load_to_lds(g0, g1, z4, z4, z8, 0);
}

__device__ __forceinline__ unsigned lds_addr_of(const void* p) {
  // flat shared address: low 32 bits are the LDS byte offset
  return (unsigned)(size_t)p;
}

// ---------------------------------------------------------------------------
// 1) LayerNorm (f32 stats) -> bf16 activations
// ---------------------------------------------------------------------------
__global__ void __launch_bounds__(256) ln_bf16_kernel(const float* __restrict__ x,
                                                      const float* __restrict__ gamma,
                                                      const float* __restrict__ beta,
                                                      bf16_t* __restrict__ xn) {
  const int row = blockIdx.x;
  const float* xr = x + (size_t)row * DIM;
  float s = 0.f, s2 = 0.f;
  for (int i = threadIdx.x; i < DIM; i += 256) {
    float vv = xr[i];
    s += vv;
    s2 += vv * vv;
  }
#pragma unroll
  for (int off = 16; off >= 1; off >>= 1) {
    s  += __shfl_xor(s,  off, 32);
    s2 += __shfl_xor(s2, off, 32);
  }
  __shared__ float ws1[8], ws2[8];
  const int wid = threadIdx.x >> 5, lid = threadIdx.x & 31;
  if (lid == 0) { ws1[wid] = s; ws2[wid] = s2; }
  __syncthreads();
  if (wid == 0) {
    s  = (lid < 8) ? ws1[lid] : 0.f;
    s2 = (lid < 8) ? ws2[lid] : 0.f;
#pragma unroll
    for (int off = 4; off >= 1; off >>= 1) {
      s  += __shfl_xor(s,  off, 32);
      s2 += __shfl_xor(s2, off, 32);
    }
    if (lid == 0) { ws1[0] = s; ws2[0] = s2; }
  }
  __syncthreads();
  const float mean = ws1[0] * (1.f / DIM);
  const float var  = ws2[0] * (1.f / DIM) - mean * mean;
  const float rstd = rsqrtf(var + 1e-5f);
  bf16_t* xo = xn + (size_t)row * DIM;
  for (int i = threadIdx.x; i < DIM; i += 256) {
    float vv = (xr[i] - mean) * rstd * gamma[i] + beta[i];
    xo[i] = (bf16_t)vv;
  }
}

// ---------------------------------------------------------------------------
// 2) fp32 -> bf16 conversion (weights)
// ---------------------------------------------------------------------------
__global__ void __launch_bounds__(256) f32_to_bf16_kernel(const float* __restrict__ src,
                                                          bf16_t* __restrict__ dst, int n) {
  int i = blockIdx.x * 256 + threadIdx.x;
  const int stride = gridDim.x * 256;
  for (; i < n; i += stride) dst[i] = (bf16_t)src[i];
}

// ---------------------------------------------------------------------------
// 3) QKV GEMM: [4096,1024] x [1024,3072]; 128x128 tile, BK=32, double-buffered
//    LDS, B stored transposed so all fragment reads are ds_load_b128 pairs.
//    Epilogue scatters q/k/v to [b,h,n,d] (q pre-scaled by d^-1/2).
// ---------------------------------------------------------------------------
__global__ void __launch_bounds__(256) gemm_qkv_kernel(const bf16_t* __restrict__ A,
                                                       const bf16_t* __restrict__ B,
                                                       bf16_t* __restrict__ qb,
                                                       bf16_t* __restrict__ kb,
                                                       bf16_t* __restrict__ vb) {
  __shared__ __align__(16) bf16_t As[2][128 * 40];   // [row][k], stride 40
  __shared__ __align__(16) bf16_t Bt[2][128 * 40];   // [n][k] transposed, stride 40
  const int tid = threadIdx.x;
  const int lane = tid & 31;
  const int wave = tid >> 5;
  const int wm = wave & 3, wn = wave >> 2;
  const int gm0 = blockIdx.y * 128, gn0 = blockIdx.x * 128;

  v8f acc[2][4];
#pragma unroll
  for (int mi = 0; mi < 2; ++mi)
#pragma unroll
    for (int ni = 0; ni < 4; ++ni)
#pragma unroll
      for (int i = 0; i < 8; ++i) acc[mi][ni][i] = 0.f;

  uint2 ra[4];
  union { uint2 u; bf16_t h[4]; } rb[4];

  auto load_globals = [&](int k0) {
#pragma unroll
    for (int i = 0; i < 4; ++i) {
      int linear = (i * 256 + tid) * 4;
      int r = linear >> 5, c = linear & 31;
      ra[i] = *reinterpret_cast<const uint2*>(A + (size_t)(gm0 + r) * DIM + k0 + c);
      int r2 = linear >> 7, c2 = linear & 127;
      rb[i].u = *reinterpret_cast<const uint2*>(B + (size_t)(k0 + r2) * QKVN + gn0 + c2);
    }
  };
  auto store_lds = [&](int bufi) {
#pragma unroll
    for (int i = 0; i < 4; ++i) {
      int linear = (i * 256 + tid) * 4;
      int r = linear >> 5, c = linear & 31;
      *reinterpret_cast<uint2*>(&As[bufi][r * 40 + c]) = ra[i];
      int r2 = linear >> 7, c2 = linear & 127;
#pragma unroll
      for (int jj = 0; jj < 4; ++jj) Bt[bufi][(c2 + jj) * 40 + r2] = rb[i].h[jj];
    }
  };

  load_globals(0);
  store_lds(0);
  __syncthreads();
  int buf = 0;
  for (int k0 = 0; k0 < DIM; k0 += 32) {
    const bool pre = (k0 + 32 < DIM);
    if (pre) load_globals(k0 + 32);
    v16bf af[2], bfr[4];
#pragma unroll
    for (int mi = 0; mi < 2; ++mi)
      af[mi] = load_a_frag(As[buf], wm * 32 + mi * 16, 40, 0, lane);
#pragma unroll
    for (int ni = 0; ni < 4; ++ni)
      bfr[ni] = load_bt_frag(Bt[buf], wn * 64 + ni * 16, 40, 0, lane);
#pragma unroll
    for (int mi = 0; mi < 2; ++mi)
#pragma unroll
      for (int ni = 0; ni < 4; ++ni)
        acc[mi][ni] = wmma_bf16(af[mi], bfr[ni], acc[mi][ni]);
    if (pre) store_lds(buf ^ 1);
    __syncthreads();
    buf ^= 1;
  }

  const int nn = lane & 15, mbase = (lane >> 4) << 3;
#pragma unroll
  for (int mi = 0; mi < 2; ++mi)
#pragma unroll
    for (int ni = 0; ni < 4; ++ni) {
      int gcol = gn0 + wn * 64 + ni * 16 + nn;
      int which = gcol >> 10, rem = gcol & 1023;
      int h = rem >> 6, dd = rem & 63;
      bf16_t* dst = (which == 0) ? qb : ((which == 1) ? kb : vb);
      float sc = (which == 0) ? 0.125f : 1.f;   // q * d^-1/2
#pragma unroll
      for (int r = 0; r < 8; ++r) {
        int grow = gm0 + wm * 32 + mi * 16 + mbase + r;
        int bi = grow >> 11, si = grow & 2047;
        size_t idx = (((size_t)(bi * HEADS + h)) * SEQ + si) * DHEAD + dd;
        dst[idx] = (bf16_t)(acc[mi][ni][r] * sc);
      }
    }
}

// ---------------------------------------------------------------------------
// 4) Flash attention: grid (SEQ/128, BATCH*HEADS). Q fragments live in
//    registers; K tiles arrive via TDM (double-buffered, padded by the DMA);
//    V tiles stored transposed in LDS so P@V fragments are contiguous.
// ---------------------------------------------------------------------------
__global__ void __launch_bounds__(256) flash_attn_kernel(const bf16_t* __restrict__ q,
                                                         const bf16_t* __restrict__ k,
                                                         const bf16_t* __restrict__ v,
                                                         bf16_t* __restrict__ o) {
  __shared__ __align__(16) bf16_t Kb[2][64 * 72];   // [key][d], stride 72 (TDM pad)
  __shared__ __align__(16) bf16_t Vt[2][64 * 72];   // [d][key], stride 72
  __shared__ __align__(16) bf16_t Ps[128 * 72];     // P round-trip, stride 72
  const int tid = threadIdx.x, lane = tid & 31, wave = tid >> 5;
  const int bh = blockIdx.y;                        // b*HEADS + h
  const int qrow0 = blockIdx.x * 128;
  const size_t base = (size_t)bh * SEQ * DHEAD;
  const int nn = lane & 15, mbase = (lane >> 4) << 3;

  // Q A-fragments straight from global into registers (reused for all tiles)
  v16bf qfrag[2];
#pragma unroll
  for (int kk = 0; kk < 2; ++kk) {
    const bf16_t* p = q + base + (size_t)(qrow0 + wave * 16 + nn) * DHEAD +
                      kk * 32 + ((lane >> 4) << 3);
    v8bf lo = *reinterpret_cast<const v8bf*>(p);
    v8bf hi = *reinterpret_cast<const v8bf*>(p + 16);
    qfrag[kk] = __builtin_shufflevector(lo, hi, 0, 1, 2, 3, 4, 5, 6, 7,
                                        8, 9, 10, 11, 12, 13, 14, 15);
  }

  auto load_v_tile = [&](int j, int bufi) {
#pragma unroll
    for (int i = 0; i < 4; ++i) {
      int linear = i * 256 + tid;
      int r = linear >> 4;            // key row
      int c = (linear & 15) * 4;      // d col
      union { uint2 u; bf16_t h[4]; } val;
      val.u = *reinterpret_cast<const uint2*>(v + base + (size_t)(j + r) * DHEAD + c);
#pragma unroll
      for (int jj = 0; jj < 4; ++jj) Vt[bufi][(c + jj) * 72 + r] = val.h[jj];
    }
  };

  float mrun[8], lrun[8];
  v8f oacc[4];
#pragma unroll
  for (int r = 0; r < 8; ++r) { mrun[r] = -1e30f; lrun[r] = 0.f; }
#pragma unroll
  for (int c = 0; c < 4; ++c)
#pragma unroll
    for (int i = 0; i < 8; ++i) oacc[c][i] = 0.f;

  // Prologue: tile 0
  if (wave == 0) tdm_load_tile64(k + base, lds_addr_of(&Kb[0][0]));
  load_v_tile(0, 0);
  if (wave == 0) __builtin_amdgcn_s_wait_tensorcnt(0);
  __syncthreads();

  int buf = 0;
  for (int j = 0; j < SEQ; j += 64) {
    const bool pre = (j + 64 < SEQ);
    if (pre) {
      if (wave == 0)
        tdm_load_tile64(k + base + (size_t)(j + 64) * DHEAD,
                        lds_addr_of(&Kb[buf ^ 1][0]));
      load_v_tile(j + 64, buf ^ 1);
    }

    // S[16x64] = Q @ K^T (q pre-scaled); K^T frags are contiguous row reads
    v8f sacc[4];
#pragma unroll
    for (int c = 0; c < 4; ++c)
#pragma unroll
      for (int i = 0; i < 8; ++i) sacc[c][i] = 0.f;
#pragma unroll
    for (int kk = 0; kk < 2; ++kk)
#pragma unroll
      for (int c = 0; c < 4; ++c) {
        v16bf b = load_bt_frag(Kb[buf], c * 16, 72, kk * 32, lane);
        sacc[c] = wmma_bf16(qfrag[kk], b, sacc[c]);
      }

    // Online softmax: per-row reductions across 16 lanes (xor 1/2/4/8)
    float alpha[8];
#pragma unroll
    for (int r = 0; r < 8; ++r) {
      float mx = sacc[0][r];
#pragma unroll
      for (int c = 1; c < 4; ++c) mx = fmaxf(mx, sacc[c][r]);
      mx = fmaxf(mx, __shfl_xor(mx, 1, 32));
      mx = fmaxf(mx, __shfl_xor(mx, 2, 32));
      mx = fmaxf(mx, __shfl_xor(mx, 4, 32));
      mx = fmaxf(mx, __shfl_xor(mx, 8, 32));
      float mnew = fmaxf(mrun[r], mx);
      alpha[r] = __expf(mrun[r] - mnew);
      mrun[r] = mnew;
    }
    float rsum[8];
#pragma unroll
    for (int r = 0; r < 8; ++r) rsum[r] = 0.f;
#pragma unroll
    for (int c = 0; c < 4; ++c)
#pragma unroll
      for (int r = 0; r < 8; ++r) {
        float p = __expf(sacc[c][r] - mrun[r]);
        sacc[c][r] = p;
        rsum[r] += p;
      }
#pragma unroll
    for (int r = 0; r < 8; ++r) {
      rsum[r] += __shfl_xor(rsum[r], 1, 32);
      rsum[r] += __shfl_xor(rsum[r], 2, 32);
      rsum[r] += __shfl_xor(rsum[r], 4, 32);
      rsum[r] += __shfl_xor(rsum[r], 8, 32);
      lrun[r] = lrun[r] * alpha[r] + rsum[r];
#pragma unroll
      for (int c = 0; c < 4; ++c) oacc[c][r] *= alpha[r];
    }

    // P (C-layout f32) -> LDS bf16 (own 16 rows; same-wave LDS order suffices)
#pragma unroll
    for (int c = 0; c < 4; ++c)
#pragma unroll
      for (int r = 0; r < 8; ++r)
        Ps[(wave * 16 + mbase + r) * 72 + c * 16 + nn] = (bf16_t)sacc[c][r];

    // O += P[16x64] @ V[64x64]; V frags contiguous from transposed tile
#pragma unroll
    for (int kk = 0; kk < 2; ++kk) {
      v16bf a = load_a_frag(Ps, wave * 16, 72, kk * 32, lane);
#pragma unroll
      for (int c = 0; c < 4; ++c) {
        v16bf b = load_bt_frag(Vt[buf], c * 16, 72, kk * 32, lane);
        oacc[c] = wmma_bf16(a, b, oacc[c]);
      }
    }

    if (pre && wave == 0) __builtin_amdgcn_s_wait_tensorcnt(0);
    __syncthreads();
    buf ^= 1;
  }

  // Normalize and store attn-out [b*n][h*d] (bf16 A for the final GEMM)
  const int hi = bh & (HEADS - 1);
  const int bi = bh >> 4;
#pragma unroll
  for (int c = 0; c < 4; ++c)
#pragma unroll
    for (int r = 0; r < 8; ++r) {
      float inv = 1.f / lrun[r];
      int grow = bi * SEQ + qrow0 + wave * 16 + mbase + r;
      int gcol = hi * DHEAD + c * 16 + nn;
      o[(size_t)grow * DIM + gcol] = (bf16_t)(oacc[c][r] * inv);
    }
}

// ---------------------------------------------------------------------------
// 5) Output GEMM: [4096,1024] x [1024,1024] + bias -> f32 d_out
// ---------------------------------------------------------------------------
__global__ void __launch_bounds__(256) gemm_out_kernel(const bf16_t* __restrict__ A,
                                                       const bf16_t* __restrict__ B,
                                                       const float* __restrict__ bias,
                                                       float* __restrict__ C) {
  __shared__ __align__(16) bf16_t As[2][128 * 40];
  __shared__ __align__(16) bf16_t Bt[2][128 * 40];
  const int tid = threadIdx.x;
  const int lane = tid & 31;
  const int wave = tid >> 5;
  const int wm = wave & 3, wn = wave >> 2;
  const int gm0 = blockIdx.y * 128, gn0 = blockIdx.x * 128;

  v8f acc[2][4];
#pragma unroll
  for (int mi = 0; mi < 2; ++mi)
#pragma unroll
    for (int ni = 0; ni < 4; ++ni)
#pragma unroll
      for (int i = 0; i < 8; ++i) acc[mi][ni][i] = 0.f;

  uint2 ra[4];
  union { uint2 u; bf16_t h[4]; } rb[4];

  auto load_globals = [&](int k0) {
#pragma unroll
    for (int i = 0; i < 4; ++i) {
      int linear = (i * 256 + tid) * 4;
      int r = linear >> 5, c = linear & 31;
      ra[i] = *reinterpret_cast<const uint2*>(A + (size_t)(gm0 + r) * DIM + k0 + c);
      int r2 = linear >> 7, c2 = linear & 127;
      rb[i].u = *reinterpret_cast<const uint2*>(B + (size_t)(k0 + r2) * DIM + gn0 + c2);
    }
  };
  auto store_lds = [&](int bufi) {
#pragma unroll
    for (int i = 0; i < 4; ++i) {
      int linear = (i * 256 + tid) * 4;
      int r = linear >> 5, c = linear & 31;
      *reinterpret_cast<uint2*>(&As[bufi][r * 40 + c]) = ra[i];
      int r2 = linear >> 7, c2 = linear & 127;
#pragma unroll
      for (int jj = 0; jj < 4; ++jj) Bt[bufi][(c2 + jj) * 40 + r2] = rb[i].h[jj];
    }
  };

  load_globals(0);
  store_lds(0);
  __syncthreads();
  int buf = 0;
  for (int k0 = 0; k0 < DIM; k0 += 32) {
    const bool pre = (k0 + 32 < DIM);
    if (pre) load_globals(k0 + 32);
    v16bf af[2], bfr[4];
#pragma unroll
    for (int mi = 0; mi < 2; ++mi)
      af[mi] = load_a_frag(As[buf], wm * 32 + mi * 16, 40, 0, lane);
#pragma unroll
    for (int ni = 0; ni < 4; ++ni)
      bfr[ni] = load_bt_frag(Bt[buf], wn * 64 + ni * 16, 40, 0, lane);
#pragma unroll
    for (int mi = 0; mi < 2; ++mi)
#pragma unroll
      for (int ni = 0; ni < 4; ++ni)
        acc[mi][ni] = wmma_bf16(af[mi], bfr[ni], acc[mi][ni]);
    if (pre) store_lds(buf ^ 1);
    __syncthreads();
    buf ^= 1;
  }

  const int nn = lane & 15, mbase = (lane >> 4) << 3;
#pragma unroll
  for (int mi = 0; mi < 2; ++mi)
#pragma unroll
    for (int ni = 0; ni < 4; ++ni)
#pragma unroll
      for (int r = 0; r < 8; ++r) {
        int grow = gm0 + wm * 32 + mi * 16 + mbase + r;
        int gcol = gn0 + wn * 64 + ni * 16 + nn;
        C[(size_t)grow * DIM + gcol] = acc[mi][ni][r] + bias[gcol];
      }
}

// ---------------------------------------------------------------------------
// Launch
// ---------------------------------------------------------------------------
extern "C" void kernel_launch(void* const* d_in, const int* in_sizes, int n_in,
                              void* d_out, int out_size, void* d_ws, size_t ws_size,
                              hipStream_t stream) {
  (void)in_sizes; (void)n_in; (void)out_size; (void)ws_size;
  const float* x      = (const float*)d_in[0];
  const float* gamma  = (const float*)d_in[1];
  const float* beta   = (const float*)d_in[2];
  const float* w_qkv  = (const float*)d_in[3];
  const float* w_out  = (const float*)d_in[4];
  const float* b_out  = (const float*)d_in[5];
  float* out = (float*)d_out;

  char* ws = (char*)d_ws;
  bf16_t* xn      = (bf16_t*)ws; ws += (size_t)ROWS * DIM * 2;       // 8 MB
  bf16_t* wqkv_bf = (bf16_t*)ws; ws += (size_t)DIM * QKVN * 2;       // 6 MB
  bf16_t* wout_bf = (bf16_t*)ws; ws += (size_t)DIM * DIM * 2;        // 2 MB
  bf16_t* qb      = (bf16_t*)ws; ws += (size_t)ROWS * DIM * 2;       // 8 MB
  bf16_t* kb      = (bf16_t*)ws; ws += (size_t)ROWS * DIM * 2;       // 8 MB
  bf16_t* vb      = (bf16_t*)ws; ws += (size_t)ROWS * DIM * 2;       // 8 MB
  bf16_t* attn    = (bf16_t*)ws; ws += (size_t)ROWS * DIM * 2;       // 8 MB

  ln_bf16_kernel<<<dim3(ROWS), dim3(256), 0, stream>>>(x, gamma, beta, xn);
  f32_to_bf16_kernel<<<dim3(2048), dim3(256), 0, stream>>>(w_qkv, wqkv_bf, DIM * QKVN);
  f32_to_bf16_kernel<<<dim3(1024), dim3(256), 0, stream>>>(w_out, wout_bf, DIM * DIM);
  gemm_qkv_kernel<<<dim3(QKVN / 128, ROWS / 128), dim3(256), 0, stream>>>(xn, wqkv_bf,
                                                                          qb, kb, vb);
  flash_attn_kernel<<<dim3(SEQ / 128, BATCH * HEADS), dim3(256), 0, stream>>>(qb, kb, vb,
                                                                              attn);
  gemm_out_kernel<<<dim3(DIM / 128, ROWS / 128), dim3(256), 0, stream>>>(attn, wout_bf,
                                                                         b_out, out);
}
